// GraphConvolutionBS_ortho_68247030334289
// MI455X (gfx1250) — matrix operand order repaired and checked by reading, not compiled, not run on hardware
//
#include <hip/hip_runtime.h>
#include <hip/hip_bf16.h>
#include <stdint.h>

#define D        512
#define NS_STEPS 5
#define BETA_W   0.5f
#define EPS_ORTH 1e-5f
#define EPS_BN   1e-5f

typedef __attribute__((ext_vector_type(16))) __bf16 v16bf;
typedef __attribute__((ext_vector_type(8)))  float  v8f;
typedef __attribute__((ext_vector_type(2)))  float  v2f;

// ---------------------------------------------------------------------------
// Phase A helpers (512x512 ortho branch)
// ---------------------------------------------------------------------------

// we = BETA*W + (1-BETA)*I, then subtract row mean -> Zc.  One block per row.
__global__ void __launch_bounds__(256) prep_zc(const float* __restrict__ w,
                                               float* __restrict__ zc) {
  __shared__ float red[256];
  const int row = blockIdx.x;
  const int t   = threadIdx.x;
  float v0 = BETA_W * w[row * D + t]       + ((row == t)       ? (1.0f - BETA_W) : 0.0f);
  float v1 = BETA_W * w[row * D + t + 256] + ((row == t + 256) ? (1.0f - BETA_W) : 0.0f);
  red[t] = v0 + v1;
  __syncthreads();
  for (int s = 128; s > 0; s >>= 1) {
    if (t < s) red[t] += red[t + s];
    __syncthreads();
  }
  const float mean = red[0] * (1.0f / (float)D);
  zc[row * D + t]       = v0 - mean;
  zc[row * D + t + 256] = v1 - mean;
}

__global__ void init_stats(float* sumsq, float* colsum, float* colsumsq) {
  const int t = threadIdx.x;
  if (t == 0) *sumsq = 0.0f;
  if (t < D) { colsum[t] = 0.0f; colsumsq[t] = 0.0f; }
}

__global__ void __launch_bounds__(256) sumsq_reduce(const float* __restrict__ s,
                                                    float* __restrict__ out) {
  __shared__ float red[256];
  const int t = threadIdx.x;
  float acc = 0.0f;
  for (int i = blockIdx.x * 256 + t; i < D * D; i += gridDim.x * 256) {
    const float v = s[i];
    acc += v * v;
  }
  red[t] = acc;
  __syncthreads();
  for (int k = 128; k > 0; k >>= 1) {
    if (t < k) red[t] += red[t + k];
    __syncthreads();
  }
  if (t == 0) atomicAdd(out, red[0]);
}

// S /= sqrt(sumsq)   (i.e. divide by Frobenius norm)
__global__ void scale_by_invnorm(float* __restrict__ s, const float* __restrict__ sumsq) {
  const int i = blockIdx.x * blockDim.x + threadIdx.x;
  if (i < D * D) s[i] *= rsqrtf(*sumsq);
}

__global__ void set_eye(float* __restrict__ b) {
  const int i = blockIdx.x * blockDim.x + threadIdx.x;
  if (i < D * D) b[i] = ((i / D) == (i % D)) ? 1.0f : 0.0f;
}

// ---------------------------------------------------------------------------
// FP32 512x512x512 GEMM: Dout = alpha*A@op(B) + beta*Cin (+eps on diagonal,
// optionally * 1/sqrt(sqrt(*qrtScale))).  One wave per 16x16 tile, K via
// V_WMMA_F32_16X16X4_F32.  TRANSB / QSCALE are compile-time so the inner
// loop is straight-line: 2x global_load_b64 + v_wmma per K-step, unrolled x4.
// ---------------------------------------------------------------------------
template <bool TRANSB, bool QSCALE>
__global__ void __launch_bounds__(32)
gemm512_f32(const float* __restrict__ A, const float* __restrict__ B,
            const float* __restrict__ Cin, float* __restrict__ Dout,
            float alpha, float beta, float diagEps,
            const float* __restrict__ qrtScale) {
  const int lane  = threadIdx.x & 31;
  const int tileN = blockIdx.x;   // 0..31
  const int tileM = blockIdx.y;   // 0..31
  const int mlo   = lane & 15;
  const int hi    = (lane >> 4) & 1;   // lane group
  float res[8];
#if __has_builtin(__builtin_amdgcn_wmma_f32_16x16x4_f32)
  v8f acc = {};
  const float* arow = A + (tileM * 16 + mlo) * D + hi * 2;
  const float* bp   = TRANSB ? (B + (tileN * 16 + mlo) * D + hi * 2)
                             : (B + (hi * 2) * D + tileN * 16 + mlo);
#pragma unroll 1
  for (int k0 = 0; k0 < D; k0 += 16) {
#pragma unroll
    for (int kk = 0; kk < 16; kk += 4) {
      v2f a, b;
      a.x = arow[k0 + kk];
      a.y = arow[k0 + kk + 1];
      if (TRANSB) {
        b.x = bp[k0 + kk];
        b.y = bp[k0 + kk + 1];
      } else {
        b.x = bp[(k0 + kk) * D];
        b.y = bp[(k0 + kk) * D + D];
      }
      acc = __builtin_amdgcn_wmma_f32_16x16x4_f32(false, a, false, b,
                                                  (short)0, acc, false, false);
    }
  }
#pragma unroll
  for (int r = 0; r < 8; ++r) res[r] = acc[r];
#else
#pragma unroll
  for (int r = 0; r < 8; ++r) {
    const int row = tileM * 16 + r + hi * 8;
    const int col = tileN * 16 + mlo;
    float s = 0.0f;
    for (int k = 0; k < D; ++k)
      s += A[row * D + k] * (TRANSB ? B[col * D + k] : B[k * D + col]);
    res[r] = s;
  }
#endif
  float scale = alpha;
  if (QSCALE) scale *= rsqrtf(sqrtf(*qrtScale));  // 1/sqrt(||S||_F)
#pragma unroll
  for (int r = 0; r < 8; ++r) {
    const int row = tileM * 16 + r + hi * 8;
    const int col = tileN * 16 + mlo;
    float v = scale * res[r];
    if (beta != 0.0f) v += beta * Cin[row * D + col];
    if (row == col)   v += diagEps;
    Dout[row * D + col] = v;
  }
}

// ---------------------------------------------------------------------------
// fp32 -> bf16 conversion
// ---------------------------------------------------------------------------
__global__ void f32_to_bf16(const float* __restrict__ in, __bf16* __restrict__ out,
                            int n) {
  const int i = blockIdx.x * blockDim.x + threadIdx.x;
  if (i < n) out[i] = (__bf16)in[i];
}

// ---------------------------------------------------------------------------
// Big bf16 GEMM: Dout[M x 512] = A[M x 512] @ Bm[512 x 512], f32 accumulate.
// 256 threads = 8 waves; block tile 128(M) x 64(N); wave tile 32x32 = 2x2 WMMA.
// B is staged into LDS pre-packed in WMMA K-pair fragment order so each lane
// reads its fragment as 8x ds_load_b32 (no b16 repacking moves).  Row stride
// of 66 dwords offsets hi-group reads by 264 dwords (4 k-pair rows), so bank sets overlap.
// ---------------------------------------------------------------------------
__global__ void __launch_bounds__(256)
gemm_bf16_big(const __bf16* __restrict__ A, const __bf16* __restrict__ Bm,
              float* __restrict__ Dout) {
  // 64-K slice: 32 k-pairs x (64 cols + 2 pad) dwords = 8448 B
  __shared__ uint32_t Bsp[32 * 66];
  const int tid    = threadIdx.x;
  const int lane   = tid & 31;
  const int wave   = tid >> 5;
  const int waveM  = wave & 3;     // 0..3
  const int waveN  = wave >> 2;    // 0..1
  const int blockM = blockIdx.y * 128;
  const int blockN = blockIdx.x * 64;
  const int mlo    = lane & 15;
  const int hi     = lane >> 4;    // 0 or 1

  v8f zero = {};
  v8f acc[2][2];
  acc[0][0] = zero; acc[0][1] = zero; acc[1][0] = zero; acc[1][1] = zero;

  for (int k0 = 0; k0 < D; k0 += 64) {
    __syncthreads();  // protect previous iteration's LDS reads
    {
      // stage B[k0..k0+63, blockN..blockN+63], packing K-pairs per column:
      // Bsp[kp*66 + n] = { B[k0+2kp, blockN+n], B[k0+2kp+1, blockN+n] }
      const uint32_t* srcu = (const uint32_t*)(Bm + (size_t)k0 * D + blockN);
#pragma unroll
      for (int p = 0; p < 4; ++p) {
        const int u  = tid + p * 256;  // 0..1023 units
        const int kp = u >> 5;         // k-pair 0..31
        const int np = u & 31;         // n-pair 0..31
        const uint32_t u0 = srcu[(2 * kp)     * (D / 2) + np];  // row k,   cols 2np..2np+1
        const uint32_t u1 = srcu[(2 * kp + 1) * (D / 2) + np];  // row k+1, cols 2np..2np+1
        Bsp[kp * 66 + 2 * np]     = __builtin_amdgcn_perm(u1, u0, 0x05040100u);
        Bsp[kp * 66 + 2 * np + 1] = __builtin_amdgcn_perm(u1, u0, 0x07060302u);
      }
    }
    __syncthreads();

#pragma unroll
    for (int ks = 0; ks < 2; ++ks) {           // two 32-K sub-steps
      const int kb0 = ks * 32;
      // A fragments: row = lane%16; VGPR q holds K pair per documented striping
      union { v16bf v; uint32_t u[8]; } afr[2];
#pragma unroll
      for (int tm = 0; tm < 2; ++tm) {
        const int row = blockM + waveM * 32 + tm * 16 + mlo;
        const uint32_t* arow = (const uint32_t*)(A + (size_t)row * D + k0 + kb0);
#pragma unroll
        for (int q = 0; q < 8; ++q) {
          const int kb = (hi ? 8 : 0) + ((q >= 4) ? 16 : 0) + ((q & 3) * 2);
          afr[tm].u[q] = arow[kb >> 1];
        }
      }
      // B fragments: u[e2] = packed K-pair (8*hi + 16*(e2>=4) + 2*(e2&3), +1) of column ncol
      union { v16bf v; uint32_t u[8]; } bfr[2];
#pragma unroll
      for (int tn = 0; tn < 2; ++tn) {
        const int ncol = waveN * 32 + tn * 16 + mlo;
#pragma unroll
        for (int e2 = 0; e2 < 8; ++e2) {
          const int kp = ks * 16 + hi * 4 + ((e2 >= 4) ? 8 : 0) + (e2 & 3);
          bfr[tn].u[e2] = Bsp[kp * 66 + ncol];
        }
      }
#pragma unroll
      for (int tm = 0; tm < 2; ++tm)
#pragma unroll
        for (int tn = 0; tn < 2; ++tn)
          acc[tm][tn] = __builtin_amdgcn_wmma_f32_16x16x32_bf16(
              false, afr[tm].v, false, bfr[tn].v, (short)0, acc[tm][tn],
              false, false);
    }
  }

  // D layout: VGPR r -> M = r + 8*hi, N = lane%16
#pragma unroll
  for (int tm = 0; tm < 2; ++tm)
#pragma unroll
    for (int tn = 0; tn < 2; ++tn)
#pragma unroll
      for (int r = 0; r < 8; ++r) {
        const int row = blockM + waveM * 32 + tm * 16 + r + hi * 8;
        const int col = blockN + waveN * 32 + tn * 16 + mlo;
        Dout[(size_t)row * D + col] = acc[tm][tn][r];
      }
}

// ---------------------------------------------------------------------------
// SpMM scatter: out[dst] += val * support[src].  One 128-thread block per edge.
// ---------------------------------------------------------------------------
__global__ void __launch_bounds__(128)
spmm_scatter(const int* __restrict__ src, const int* __restrict__ dst,
             const float* __restrict__ val, const float* __restrict__ support,
             float* __restrict__ out, int nE) {
  const int e = blockIdx.x;
  if (e >= nE) return;
  const int   s = src[e];
  const int   d = dst[e];
  const float w = val[e];
  const int   t = threadIdx.x;
  if (e + 16 < nE) {  // gfx1250 global_prefetch of an upcoming gather row
    const int sf = src[e + 16];
    __builtin_prefetch(support + (size_t)sf * D + t * 4, 0, 0);
  }
  const float4 m = ((const float4*)(support + (size_t)s * D))[t];
  float* orow = out + (size_t)d * D + t * 4;
  atomicAdd(orow + 0, w * m.x);
  atomicAdd(orow + 1, w * m.y);
  atomicAdd(orow + 2, w * m.z);
  atomicAdd(orow + 3, w * m.w);
}

// ---------------------------------------------------------------------------
// BatchNorm: per-column sum / sum-of-squares, then normalize + ReLU in place.
// ---------------------------------------------------------------------------
__global__ void __launch_bounds__(512)
bn_stats(const float* __restrict__ out, float* __restrict__ colsum,
         float* __restrict__ colsumsq, int nRows, int rowsPerBlock) {
  const int c  = threadIdx.x;               // column 0..511
  const int r0 = blockIdx.x * rowsPerBlock;
  int r1 = r0 + rowsPerBlock;
  if (r1 > nRows) r1 = nRows;
  float s = 0.0f, q = 0.0f;
  for (int r = r0; r < r1; ++r) {
    const float v = out[(size_t)r * D + c];
    s += v;
    q += v * v;
  }
  atomicAdd(&colsum[c], s);
  atomicAdd(&colsumsq[c], q);
}

__global__ void __launch_bounds__(256)
bn_apply(float* __restrict__ out, const float* __restrict__ colsum,
         const float* __restrict__ colsumsq, const float* __restrict__ gamma,
         const float* __restrict__ beta, int nRows) {
  const int i = blockIdx.x * blockDim.x + threadIdx.x;  // float4 index
  const int total = nRows * (D / 4);
  if (i >= total) return;
  const int   cb   = (i % (D / 4)) * 4;
  const float invN = 1.0f / (float)nRows;
  float4* p = ((float4*)out) + i;
  float4 v = *p;
  float vv[4] = {v.x, v.y, v.z, v.w};
  float rr[4];
#pragma unroll
  for (int j = 0; j < 4; ++j) {
    const int   c    = cb + j;
    const float mean = colsum[c] * invN;
    const float var  = colsumsq[c] * invN - mean * mean;
    float r = (vv[j] - mean) * rsqrtf(var + EPS_BN) * gamma[c] + beta[c];
    rr[j] = r > 0.0f ? r : 0.0f;
  }
  *p = make_float4(rr[0], rr[1], rr[2], rr[3]);
}

// ---------------------------------------------------------------------------
// Orchestration
// ---------------------------------------------------------------------------
extern "C" void kernel_launch(void* const* d_in, const int* in_sizes, int n_in,
                              void* d_out, int out_size, void* d_ws, size_t ws_size,
                              hipStream_t stream) {
  (void)n_in; (void)out_size; (void)ws_size;
  const float* x      = (const float*)d_in[0];
  const int*   esrc   = (const int*)d_in[1];
  const int*   edst   = (const int*)d_in[2];
  const float* evalp  = (const float*)d_in[3];
  const float* weight = (const float*)d_in[4];
  const float* selfw  = (const float*)d_in[5];
  const float* gamma  = (const float*)d_in[6];
  const float* betap  = (const float*)d_in[7];
  const int N  = in_sizes[0] / D;
  const int nE = in_sizes[1];

  char* ws = (char*)d_ws;
  const size_t MB = 1u << 20;
  float*  Zc       = (float*)(ws + 0 * MB);
  float*  S        = (float*)(ws + 1 * MB);
  float*  B0       = (float*)(ws + 2 * MB);
  float*  B1       = (float*)(ws + 3 * MB);
  float*  T1       = (float*)(ws + 4 * MB);
  float*  T2       = (float*)(ws + 5 * MB);
  float*  tF       = (float*)(ws + 6 * MB);
  __bf16* tB       = (__bf16*)(ws + 7 * MB);
  __bf16* swB      = (__bf16*)(ws + 7 * MB + 512 * 1024);
  float*  sumsq    = (float*)(ws + 8 * MB);
  float*  colsum   = (float*)(ws + 8 * MB + 1024);
  float*  colsumsq = (float*)(ws + 8 * MB + 1024 + 2048);
  __bf16* xB       = (__bf16*)(ws + 8 * MB + 16 * 1024);
  float*  support  = (float*)(ws + 8 * MB + 16 * 1024 + (size_t)N * D * sizeof(__bf16));
  float*  outp     = (float*)d_out;

  const dim3 g32(32, 32);

  // --- Phase A: ortho transform on 512x512 (fp32 WMMA) ---
  prep_zc<<<D, 256, 0, stream>>>(weight, Zc);
  init_stats<<<1, 512, 0, stream>>>(sumsq, colsum, colsumsq);
  // S = Zc @ Zc^T + eps*I
  gemm512_f32<true, false><<<g32, 32, 0, stream>>>(Zc, Zc, Zc, S, 1.0f, 0.0f,
                                                   EPS_ORTH, nullptr);
  sumsq_reduce<<<256, 256, 0, stream>>>(S, sumsq);
  scale_by_invnorm<<<(D * D + 255) / 256, 256, 0, stream>>>(S, sumsq);
  set_eye<<<(D * D + 255) / 256, 256, 0, stream>>>(B0);
  float* Bc = B0;
  float* Bn = B1;
  for (int it = 0; it < NS_STEPS; ++it) {
    gemm512_f32<false, false><<<g32, 32, 0, stream>>>(Bc, Bc, Bc, T1, 1.0f, 0.0f, 0.0f, nullptr);
    gemm512_f32<false, false><<<g32, 32, 0, stream>>>(T1, Bc, Bc, T2, 1.0f, 0.0f, 0.0f, nullptr);
    // Bn = 1.5*Bc - 0.5*(T2 @ S)
    gemm512_f32<false, false><<<g32, 32, 0, stream>>>(T2, S, Bc, Bn, -0.5f, 1.5f, 0.0f, nullptr);
    float* tmp = Bc; Bc = Bn; Bn = tmp;
  }
  // t = (B @ Zc) / sqrt(norm_S),  norm_S = sqrt(sumsq)
  gemm512_f32<false, true><<<g32, 32, 0, stream>>>(Bc, Zc, Zc, tF, 1.0f, 0.0f, 0.0f, sumsq);

  // --- conversions to bf16 ---
  f32_to_bf16<<<(D * D + 255) / 256, 256, 0, stream>>>(tF, tB, D * D);
  f32_to_bf16<<<(D * D + 255) / 256, 256, 0, stream>>>(selfw, swB, D * D);
  f32_to_bf16<<<(N * D + 255) / 256, 256, 0, stream>>>(x, xB, N * D);

  // --- Phase B: big bf16 WMMA GEMMs ---
  dim3 gg(D / 64, N / 128);
  gemm_bf16_big<<<gg, 256, 0, stream>>>(xB, tB, support);   // support = x @ t
  gemm_bf16_big<<<gg, 256, 0, stream>>>(xB, swB, outp);     // out = x @ self_weight

  // --- SpMM scatter-add into out ---
  spmm_scatter<<<nE, 128, 0, stream>>>(esrc, edst, evalp, support, outp, nE);

  // --- BatchNorm + ReLU ---
  const int rowsPerBlock = 256;
  bn_stats<<<(N + rowsPerBlock - 1) / rowsPerBlock, 512, 0, stream>>>(
      outp, colsum, colsumsq, N, rowsPerBlock);
  bn_apply<<<(N * (D / 4) + 255) / 256, 256, 0, stream>>>(
      outp, colsum, colsumsq, gamma, betap, N);
}
